// Seq2SeqRNN_89481348645592
// MI455X (gfx1250) — matrix operand order repaired and checked by reading, not compiled, not run on hardware
//
#include <hip/hip_runtime.h>

// ---------------- problem constants ----------------
#define BB   512
#define LIN_ 96
#define LOUT_ 96
#define NUM_ 64
#define EMB_ 32
#define HID_ 256
#define IN_  96
#define G3_  768
#define TT   (LIN_ + LOUT_)   // 192

// fragment geometry (f16 WMMA 16x16x32): one B-frag = 32 lanes * 16 halves = 512 halves
#define FRAG_H 512
#define NT_G   48            // 768/16 gate N-tiles
#define KT_X   3             // 96/32
#define KT_H   8             // 256/32
#define NT_D   4             // 64/16

// workspace layout (in _Float16 units)
#define OFF_WIH  ((size_t)0)
#define OFF_WHH  (OFF_WIH + (size_t)NT_G*KT_X*FRAG_H)            // 144 frags
#define OFF_WDEC (OFF_WHH + (size_t)NT_G*KT_H*FRAG_H)            // 384 frags
#define OFF_XENC (OFF_WDEC + (size_t)NT_D*KT_H*FRAG_H)           // 32 frags
#define OFF_EDEC (OFF_XENC + (size_t)BB*LIN_*IN_)                // enc x f16
// total = OFF_EDEC + BB*LOUT_*EMB_  (~13.2 MB)

#define HS 264   // LDS stride (halves) for h rows: 528B -> conflict-light
#define XS 104   // LDS stride for x rows

typedef __attribute__((ext_vector_type(16))) _Float16 v16h;
typedef __attribute__((ext_vector_type(8)))  _Float16 v8h;
typedef __attribute__((ext_vector_type(8)))  float    v8f;

__device__ __forceinline__ v8f zero8() {
  v8f z;
#pragma unroll
  for (int i = 0; i < 8; ++i) z[i] = 0.f;
  return z;
}

// A-fragment (16x32 f16) from a row-major LDS tile, per ISA 16-bit A layout:
// lane L: M=L&15; halves 0..7 -> K = g*8+hv ; halves 8..15 -> K = 16+g*8+(hv-8); g=L>>4
__device__ __forceinline__ v16h load_a(const _Float16* rowbase, int stride, int lane, int koff) {
  const int m = lane & 15, g = lane >> 4;
  const _Float16* p = rowbase + m * stride + koff + g * 8;
  v8h lo = *(const v8h*)p;
  v8h hi = *(const v8h*)(p + 16);
  v16h a;
#pragma unroll
  for (int i = 0; i < 8; ++i) { a[i] = lo[i]; a[8 + i] = hi[i]; }
  return a;
}

__device__ __forceinline__ float sigmoidf_(float x) { return 1.f / (1.f + __expf(-x)); }
__device__ __forceinline__ float tanhf_(float x)    { return 1.f - 2.f / (__expf(2.f * x) + 1.f); }

// ---------------- prep kernel 1: swizzle weights into WMMA B-fragments ----------------
// B-frag layout (32x16 f16): lane L: N = nt*16 + (L&15); half hv<8 -> K = kt*32 + (L>>4)*8 + hv ; hv>=8 -> K = kt*32 + 16 + (L>>4)*8 + (hv-8)
__global__ void pack_frags(const float* __restrict__ W_ih, const float* __restrict__ W_hh,
                           const float* __restrict__ W_dec, _Float16* __restrict__ wsh) {
  int idx = blockIdx.x * 256 + threadIdx.x;
  const int TOTAL = (NT_G * KT_X + NT_G * KT_H + NT_D * KT_H) * 256; // u32 slots
  if (idx >= TOTAL) return;
  int f  = idx >> 8;        // global frag id
  int r  = idx & 255;       // u32 slot in frag
  int ln = r >> 3;          // lane 0..31
  int v  = r & 7;           // u32 within lane (2 halves)
  const float* W; int K; size_t base; int nt, kt, fl;
  if (f < NT_G * KT_X)            { fl = f;                 W = W_ih;  K = IN_;  base = OFF_WIH;  nt = fl / KT_X; kt = fl % KT_X; }
  else if (f < NT_G * (KT_X+KT_H)){ fl = f - NT_G * KT_X;   W = W_hh;  K = HID_; base = OFF_WHH;  nt = fl / KT_H; kt = fl % KT_H; }
  else                            { fl = f - NT_G*(KT_X+KT_H); W = W_dec; K = HID_; base = OFF_WDEC; nt = fl / KT_H; kt = fl % KT_H; }
  int n = nt * 16 + (ln & 15);
  int k = kt * 32 + (ln >> 4) * 8 + (v >> 2) * 16 + (v & 3) * 2;
  size_t o = base + (size_t)fl * FRAG_H + ln * 16 + v * 2;
  wsh[o]     = (_Float16)W[(size_t)n * K + k];
  wsh[o + 1] = (_Float16)W[(size_t)n * K + k + 1];
}

// ---------------- prep kernel 2: build f16 encoder inputs + decoder embeddings ----------------
__global__ void build_inputs(const float* __restrict__ input_num, const int* __restrict__ input_cat,
                             const float* __restrict__ emb, _Float16* __restrict__ wsh) {
  size_t idx = (size_t)blockIdx.x * 256 + threadIdx.x;
  const size_t XN = (size_t)BB * LIN_ * IN_;
  const size_t EN = (size_t)BB * LOUT_ * EMB_;
  if (idx < XN) {
    int k = (int)(idx % IN_); size_t bt = idx / IN_; int t = (int)(bt % LIN_); int b = (int)(bt / LIN_);
    float v;
    if (k < NUM_) v = input_num[((size_t)b * TT + t) * NUM_ + k];
    else          v = emb[(size_t)input_cat[(size_t)b * TT + t] * EMB_ + (k - NUM_)];
    wsh[OFF_XENC + idx] = (_Float16)v;
  } else if (idx < XN + EN) {
    size_t j = idx - XN;
    int e = (int)(j % EMB_); size_t bt = j / EMB_; int t = (int)(bt % LOUT_); int b = (int)(bt / LOUT_);
    wsh[OFF_EDEC + j] = (_Float16)emb[(size_t)input_cat[(size_t)b * TT + LIN_ + t] * EMB_ + e];
  }
}

// ---------------- main persistent GRU kernel: 32 blocks x 256 threads (8 wave32) ----------------
__global__ __launch_bounds__(256)
void seq2seq_gru(const _Float16* __restrict__ wsh,
                 const float* __restrict__ b_ih, const float* __restrict__ b_hh,
                 const float* __restrict__ b_dec, float* __restrict__ out) {
  __shared__ __align__(16) _Float16 sH[2][16 * HS];          // 16.5 KB : ping-pong hidden state
  __shared__ __align__(16) _Float16 sX[16 * XS];             //  3.3 KB : step input
  __shared__ __align__(16) _Float16 sWdec[NT_D * KT_H * FRAG_H]; // 32 KB : decoder weights

  const int tid   = threadIdx.x;
  const int lane  = tid & 31;
  const int w     = tid >> 5;        // wave 0..7, owns hidden cols [32w, 32w+32)
  const int l15   = lane & 15;
  const int bbase = blockIdx.x * 16; // batch tile

  // this wave's 6 gate N-tiles: r{2w,2w+1}, z{16+..}, n{32+..}
  int tiles[6];
  tiles[0] = 2 * w;      tiles[1] = 2 * w + 1;
  tiles[2] = 16 + 2 * w; tiles[3] = 17 + 2 * w;
  tiles[4] = 32 + 2 * w; tiles[5] = 33 + 2 * w;

  float bx[6], bh[6];
#pragma unroll
  for (int j = 0; j < 6; ++j) {
    int col = tiles[j] * 16 + l15;
    bx[j] = b_ih[col]; bh[j] = b_hh[col];
  }
  const float bd = (w < 4) ? b_dec[w * 16 + l15] : 0.f;

  float hreg[2][8];
#pragma unroll
  for (int p = 0; p < 2; ++p)
#pragma unroll
    for (int i = 0; i < 8; ++i) hreg[p][i] = 0.f;

  // one-time: zero h(0), cache W_dec fragments in LDS (16B vector copies)
  for (int i = tid; i < 16 * HS; i += 256) sH[0][i] = (_Float16)0.f;
  {
    const uint4* src = (const uint4*)(wsh + OFF_WDEC);
    uint4* dst = (uint4*)sWdec;
    const int n16 = NT_D * KT_H * FRAG_H / 8;  // 2048 uint4
    for (int i = tid; i < n16; i += 256) dst[i] = src[i];
  }
  __syncthreads();

  const _Float16* xenc = wsh + OFF_XENC + (size_t)bbase * LIN_ * IN_;
  const _Float16* edec = wsh + OFF_EDEC + (size_t)bbase * LOUT_ * EMB_;

  int cur = 0;

  // one GRU step: reads sX + sH[cur], writes sH[cur^1] and hreg
  auto gru_body = [&](int c) {
    // Opaque *integer offset* (not pointer!): prevents LICM from hoisting the 66
    // weight loads out of the time loop (which caused scratch spilling), while
    // keeping address provenance rooted at the kernel arg so the loads lower to
    // global_load_b128 (saddr form), NOT flat_load (which would tie LOADcnt to
    // DScnt and add 64-bit VGPR address math).
    unsigned off = 0;
    asm volatile("" : "+s"(off));
    const _Float16* wih = wsh + OFF_WIH + off;
    const _Float16* whh = wsh + OFF_WHH + off;

    v8f gx[6], gh[6];
#pragma unroll
    for (int j = 0; j < 6; ++j) { gx[j] = zero8(); gh[j] = zero8(); }
    // x GEMM: [16,96] x [96, 6*16]
#pragma unroll
    for (int kt = 0; kt < KT_X; ++kt) {
      v16h a = load_a(sX, XS, lane, kt * 32);
#pragma unroll
      for (int j = 0; j < 6; ++j) {
        v16h b = *(const v16h*)(wih + (size_t)(tiles[j] * KT_X + kt) * FRAG_H + lane * 16);
        gx[j] = __builtin_amdgcn_wmma_f32_16x16x32_f16(false, a, false, b, (short)0, gx[j], false, false);
      }
    }
    // h GEMM: [16,256] x [256, 6*16]
#pragma unroll
    for (int kt = 0; kt < KT_H; ++kt) {
      v16h a = load_a(&sH[c][0], HS, lane, kt * 32);
#pragma unroll
      for (int j = 0; j < 6; ++j) {
        v16h b = *(const v16h*)(whh + (size_t)(tiles[j] * KT_H + kt) * FRAG_H + lane * 16);
        gh[j] = __builtin_amdgcn_wmma_f32_16x16x32_f16(false, a, false, b, (short)0, gh[j], false, false);
      }
    }
    // gates + hidden update (this wave's 32 columns live in hreg)
    const int nxt = c ^ 1;
#pragma unroll
    for (int p = 0; p < 2; ++p) {
#pragma unroll
      for (int i = 0; i < 8; ++i) {
        float r  = sigmoidf_((gx[p][i]     + bx[p])     + (gh[p][i]     + bh[p]));
        float z  = sigmoidf_((gx[2+p][i]   + bx[2+p])   + (gh[2+p][i]   + bh[2+p]));
        float nn = tanhf_((gx[4+p][i] + bx[4+p]) + r * (gh[4+p][i] + bh[4+p]));
        float h2 = (1.f - z) * nn + z * hreg[p][i];
        hreg[p][i] = h2;
        int m = (lane >> 4) * 8 + i;
        sH[nxt][m * HS + w * 32 + p * 16 + l15] = (_Float16)h2;
      }
    }
  };

  // decoder output GEMM + stage next x into sX (waves 0-3: out; waves 4-7: emb)
  auto dec_stage = [&](int c, int t) {
    if (w < 4) {
      v8f acc = zero8();
#pragma unroll
      for (int kt = 0; kt < KT_H; ++kt) {
        v16h a = load_a(&sH[c][0], HS, lane, kt * 32);
        v16h b = *(const v16h*)(&sWdec[(size_t)(w * KT_H + kt) * FRAG_H + lane * 16]);
        acc = __builtin_amdgcn_wmma_f32_16x16x32_f16(false, a, false, b, (short)0, acc, false, false);
      }
#pragma unroll
      for (int i = 0; i < 8; ++i) {
        float v = acc[i] + bd;
        int m = (lane >> 4) * 8 + i;
        out[((size_t)(bbase + m) * LOUT_ + t) * NUM_ + w * 16 + l15] = v;
        sX[m * XS + w * 16 + l15] = (_Float16)v;   // feedback for next step
      }
    } else {
      int q0 = (w - 4) * 32 + lane;                // 0..127 over 512 emb elements
#pragma unroll
      for (int s = 0; s < 4; ++s) {
        int q = q0 + 128 * s;
        int m = q >> 5, cc = q & 31;
        sX[m * XS + 64 + cc] = edec[((size_t)m * LOUT_ + t) * EMB_ + cc];
      }
    }
  };

  // ---------------- encoder: 96 steps ----------------
  for (int t = 0; t < LIN_; ++t) {
    for (int q = tid; q < 16 * IN_ / 2; q += 256) {     // stage x tile (u32 copies)
      int m = q / 48, cc = q % 48;
      *(unsigned*)&sX[m * XS + 2 * cc] =
          *(const unsigned*)&xenc[((size_t)m * LIN_ + t) * IN_ + 2 * cc];
    }
    __syncthreads();
    gru_body(cur);
    __syncthreads();
    cur ^= 1;
  }

  // ---------------- decoder: out0 then 95 autoregressive steps ----------------
  dec_stage(cur, 0);
  __syncthreads();
  for (int t = 1; t < LOUT_; ++t) {
    gru_body(cur);
    __syncthreads();
    cur ^= 1;
    dec_stage(cur, t);
    __syncthreads();
  }
}

extern "C" void kernel_launch(void* const* d_in, const int* in_sizes, int n_in,
                              void* d_out, int out_size, void* d_ws, size_t ws_size,
                              hipStream_t stream) {
  const float* input_num = (const float*)d_in[0];
  const int*   input_cat = (const int*)  d_in[1];
  const float* emb_table = (const float*)d_in[2];
  const float* W_ih      = (const float*)d_in[3];
  const float* W_hh      = (const float*)d_in[4];
  const float* b_ih      = (const float*)d_in[5];
  const float* b_hh      = (const float*)d_in[6];
  const float* W_dec     = (const float*)d_in[7];
  const float* b_dec     = (const float*)d_in[8];
  float*       out       = (float*)d_out;
  _Float16*    wsh       = (_Float16*)d_ws;

  (void)in_sizes; (void)n_in; (void)out_size; (void)ws_size;

  // prep: weight swizzle (560 frags * 256 u32 slots)
  pack_frags<<<dim3(560), dim3(256), 0, stream>>>(W_ih, W_hh, W_dec, wsh);
  // prep: f16 encoder inputs + decoder embeddings
  const size_t nElem = (size_t)BB * LIN_ * IN_ + (size_t)BB * LOUT_ * EMB_;
  build_inputs<<<dim3((unsigned)((nElem + 255) / 256)), dim3(256), 0, stream>>>(
      input_num, input_cat, emb_table, wsh);
  // persistent batch-tiled GRU: 32 workgroups x 8 wave32
  seq2seq_gru<<<dim3(BB / 16), dim3(256), 0, stream>>>(wsh, b_ih, b_hh, b_dec, out);
}
